// GIN_31164282699923
// MI455X (gfx1250) — matrix operand order repaired and checked by reading, not compiled, not run on hardware
//
#include <hip/hip_runtime.h>

#define NN 50000
#define NE 800000
#define DCH 128
#define NPAD 50048        // 782 * 64 rows, padded for full GEMM tiles
#define GEMM_BLOCKS 782
#define STAT_BLOCKS 256

typedef __attribute__((ext_vector_type(16))) __bf16 v16bf;
typedef __attribute__((ext_vector_type(16))) float  v16f;
typedef __attribute__((ext_vector_type(8)))  float  v8f;

union BBf { v16bf v; int4 q[2]; };

// ---------------- weight prep: f32 [l][k][n] -> bf16 transposed [l][n][k] ----
__global__ void gin_prep_w(const float* __restrict__ W1, const float* __restrict__ W2,
                           unsigned short* __restrict__ Wt1, unsigned short* __restrict__ Wt2) {
  int idx = blockIdx.x * blockDim.x + threadIdx.x;
  if (idx >= 3 * DCH * DCH) return;
  int l = idx >> 14;
  int rem = idx & 16383;
  int n = rem >> 7;
  int k = rem & 127;
  size_t src = (size_t)l * DCH * DCH + (size_t)k * DCH + n;
  Wt1[idx] = __builtin_bit_cast(unsigned short, (__bf16)W1[src]);
  Wt2[idx] = __builtin_bit_cast(unsigned short, (__bf16)W2[src]);
}

// ---------------- CSR build ------------------------------------------------
__global__ void gin_zero_i(int* p, int n) {
  int i = blockIdx.x * blockDim.x + threadIdx.x;
  if (i < n) p[i] = 0;
}

__global__ void gin_hist(const int* __restrict__ dst, int* __restrict__ deg, int e) {
  int i = blockIdx.x * blockDim.x + threadIdx.x;
  if (i < e) atomicAdd(&deg[dst[i]], 1);
}

__global__ void gin_scan(const int* __restrict__ deg, int* __restrict__ rowstart,
                         int* __restrict__ cursor, int n) {
  __shared__ int lsum[1024];
  int tid = threadIdx.x;
  int ch = (n + 1023) / 1024;
  int base = tid * ch;
  int s = 0;
  for (int i = 0; i < ch; ++i) { int idx = base + i; if (idx < n) s += deg[idx]; }
  lsum[tid] = s;
  __syncthreads();
  for (int off = 1; off < 1024; off <<= 1) {
    int v = (tid >= off) ? lsum[tid - off] : 0;
    __syncthreads();
    lsum[tid] += v;
    __syncthreads();
  }
  int run = (tid == 0) ? 0 : lsum[tid - 1];
  for (int i = 0; i < ch; ++i) {
    int idx = base + i;
    if (idx < n) { rowstart[idx] = run; cursor[idx] = run; run += deg[idx]; }
  }
  if (tid == 1023) rowstart[n] = run;
}

__global__ void gin_fill(const int* __restrict__ src, const int* __restrict__ dst,
                         int* __restrict__ cursor, int* __restrict__ csr, int e) {
  int i = blockIdx.x * blockDim.x + threadIdx.x;
  if (i < e) {
    int p = atomicAdd(&cursor[dst[i]], 1);
    csr[p] = src[i];
  }
}

// sort each node's neighbor list -> deterministic summation order
__global__ void gin_sortseg(int* __restrict__ csr, const int* __restrict__ rowstart, int n) {
  int i = blockIdx.x * blockDim.x + threadIdx.x;
  if (i >= n) return;
  int s = rowstart[i], e = rowstart[i + 1];
  for (int a = s + 1; a < e; ++a) {
    int v = csr[a];
    int b = a - 1;
    while (b >= s && csr[b] > v) { csr[b + 1] = csr[b]; --b; }
    csr[b + 1] = v;
  }
}

// ---------------- aggregation: agg[i] = h[i] + sum_{j in N(i)} h[j] --------
// one wave per node; lane holds 4 channels as float4; uniform loop over CSR
__global__ void gin_gather(const float* __restrict__ hin, float* __restrict__ agg,
                           const int* __restrict__ rowstart, const int* __restrict__ csr, int n) {
  int node = (blockIdx.x * blockDim.x + threadIdx.x) >> 5;
  int lane = threadIdx.x & 31;
  if (node >= n) return;
  const float4* hv = (const float4*)hin;
  float4 acc = hv[(size_t)node * 32 + lane];
  int s = rowstart[node], e = rowstart[node + 1];
  for (int j = s; j < e; ++j) {
    float4 v = hv[(size_t)csr[j] * 32 + lane];
    acc.x += v.x; acc.y += v.y; acc.z += v.z; acc.w += v.w;
  }
  ((float4*)agg)[(size_t)node * 32 + lane] = acc;
}

// ---------------- GEMM1: Z = A @ W1 + b1 (bf16 WMMA, f32 accum) ------------
// block = 128 threads = 4 waves; wave computes 16 rows x 128 cols; K = 128
__global__ __launch_bounds__(128) void gin_gemm1(
    const float* __restrict__ A, const unsigned short* __restrict__ Wtg,
    const float* __restrict__ bias, float* __restrict__ Z) {
  __shared__ unsigned short WtL[DCH * 136];   // transposed weight, padded stride
  {
    int tid = threadIdx.x;
    int4* dv = (int4*)(&WtL[tid * 136]);
    const int4* sv = (const int4*)(Wtg + (size_t)tid * DCH);
#pragma unroll
    for (int i = 0; i < 16; ++i) dv[i] = sv[i];
  }
  __syncthreads();

  int lane = threadIdx.x & 31;
  int wv = threadIdx.x >> 5;
  int g = lane >> 4;          // lane group (selects contiguous K half, k = 16*g + s)
  int c = lane & 15;
  int rowBase = blockIdx.x * 64 + wv * 16;
  const float* arow = A + (size_t)(rowBase + c) * DCH;

  v8f zero = {};
  v8f acc[8];
#pragma unroll
  for (int t = 0; t < 8; ++t) acc[t] = zero;

#pragma unroll
  for (int kk = 0; kk < 4; ++kk) {
    int k0 = kk * 32;
    int ka = k0 + g * 16;
    int kb = ka + 8;
    float4 x0 = *(const float4*)(arow + ka);
    float4 x1 = *(const float4*)(arow + ka + 4);
    float4 x2 = *(const float4*)(arow + kb);
    float4 x3 = *(const float4*)(arow + kb + 4);
    v16f xv = {x0.x, x0.y, x0.z, x0.w, x1.x, x1.y, x1.z, x1.w,
               x2.x, x2.y, x2.z, x2.w, x3.x, x3.y, x3.z, x3.w};
    v16bf af = __builtin_convertvector(xv, v16bf);   // native pk bf16 cvt
#pragma unroll
    for (int t = 0; t < 8; ++t) {
      BBf bf;
      const int4* bp = (const int4*)(&WtL[(t * 16 + c) * 136 + k0 + g * 16]);
      bf.q[0] = bp[0];
      bf.q[1] = bp[1];
      acc[t] = __builtin_amdgcn_wmma_f32_16x16x32_bf16(false, af, false, bf.v,
                                                       (short)0, acc[t], false, false);
    }
  }

#pragma unroll
  for (int t = 0; t < 8; ++t) {
    float bv = bias[t * 16 + c];
#pragma unroll
    for (int r = 0; r < 8; ++r) {
      Z[(size_t)(rowBase + g * 8 + r) * DCH + t * 16 + c] = acc[t][r] + bv;
    }
  }
}

// ---------------- BN stats (deterministic 2-stage) -------------------------
__global__ void gin_bnstats(const float* __restrict__ Z, float* __restrict__ psum,
                            float* __restrict__ qsum, int n) {
  int c = threadIdx.x;           // 128
  int b = blockIdx.x;            // STAT_BLOCKS
  float s = 0.f, q = 0.f;
  for (int r = b; r < n; r += STAT_BLOCKS) {
    float v = Z[(size_t)r * DCH + c];
    s += v; q += v * v;
  }
  psum[b * DCH + c] = s;
  qsum[b * DCH + c] = q;
}

__global__ void gin_bnparams(const float* __restrict__ psum, const float* __restrict__ qsum,
                             const float* __restrict__ bng, const float* __restrict__ bnb,
                             float* __restrict__ scaleArr, float* __restrict__ shiftArr, int n) {
  int c = threadIdx.x;
  float s = 0.f, q = 0.f;
  for (int b = 0; b < STAT_BLOCKS; ++b) { s += psum[b * DCH + c]; q += qsum[b * DCH + c]; }
  float inv = 1.0f / (float)n;
  float mu = s * inv;
  float var = q * inv - mu * mu;
  float sc = bng[c] * rsqrtf(var + 1e-5f);
  scaleArr[c] = sc;
  shiftArr[c] = bnb[c] - mu * sc;
}

// ---------------- GEMM2: fused BN-affine+PReLU -> @W2+b2 -> LN -> PReLU ----
__global__ __launch_bounds__(128) void gin_gemm2(
    const float* __restrict__ Zin, const float* __restrict__ scaleArr,
    const float* __restrict__ shiftArr, const float* __restrict__ a1p,
    const unsigned short* __restrict__ Wtg, const float* __restrict__ bias,
    const float* __restrict__ lng, const float* __restrict__ lnb,
    const float* __restrict__ a2p, float* __restrict__ out, int nrows) {
  __shared__ unsigned short WtL[DCH * 136];
  {
    int tid = threadIdx.x;
    int4* dv = (int4*)(&WtL[tid * 136]);
    const int4* sv = (const int4*)(Wtg + (size_t)tid * DCH);
#pragma unroll
    for (int i = 0; i < 16; ++i) dv[i] = sv[i];
  }
  __syncthreads();

  float a1v = a1p[0];
  float a2v = a2p[0];
  int lane = threadIdx.x & 31;
  int wv = threadIdx.x >> 5;
  int g = lane >> 4;
  int c = lane & 15;
  int rowBase = blockIdx.x * 64 + wv * 16;
  const float* arow = Zin + (size_t)(rowBase + c) * DCH;

  v8f zero = {};
  v8f acc[8];
#pragma unroll
  for (int t = 0; t < 8; ++t) acc[t] = zero;

#pragma unroll
  for (int kk = 0; kk < 4; ++kk) {
    int k0 = kk * 32;
    int ka = k0 + g * 16;
    int kb = ka + 8;
    float4 x0 = *(const float4*)(arow + ka);
    float4 x1 = *(const float4*)(arow + ka + 4);
    float4 x2 = *(const float4*)(arow + kb);
    float4 x3 = *(const float4*)(arow + kb + 4);
    float4 s0 = *(const float4*)(scaleArr + ka);
    float4 s1 = *(const float4*)(scaleArr + ka + 4);
    float4 s2 = *(const float4*)(scaleArr + kb);
    float4 s3 = *(const float4*)(scaleArr + kb + 4);
    float4 h0 = *(const float4*)(shiftArr + ka);
    float4 h1 = *(const float4*)(shiftArr + ka + 4);
    float4 h2 = *(const float4*)(shiftArr + kb);
    float4 h3 = *(const float4*)(shiftArr + kb + 4);
    float xs[16] = {x0.x, x0.y, x0.z, x0.w, x1.x, x1.y, x1.z, x1.w,
                    x2.x, x2.y, x2.z, x2.w, x3.x, x3.y, x3.z, x3.w};
    float ss[16] = {s0.x, s0.y, s0.z, s0.w, s1.x, s1.y, s1.z, s1.w,
                    s2.x, s2.y, s2.z, s2.w, s3.x, s3.y, s3.z, s3.w};
    float hh[16] = {h0.x, h0.y, h0.z, h0.w, h1.x, h1.y, h1.z, h1.w,
                    h2.x, h2.y, h2.z, h2.w, h3.x, h3.y, h3.z, h3.w};
    v16f xv;
#pragma unroll
    for (int i = 0; i < 16; ++i) {
      float z = fmaf(xs[i], ss[i], hh[i]);   // BatchNorm affine
      xv[i] = z >= 0.f ? z : a1v * z;        // PReLU (a1)
    }
    v16bf af = __builtin_convertvector(xv, v16bf);
#pragma unroll
    for (int t = 0; t < 8; ++t) {
      BBf bf;
      const int4* bp = (const int4*)(&WtL[(t * 16 + c) * 136 + k0 + g * 16]);
      bf.q[0] = bp[0];
      bf.q[1] = bp[1];
      acc[t] = __builtin_amdgcn_wmma_f32_16x16x32_bf16(false, af, false, bf.v,
                                                       (short)0, acc[t], false, false);
    }
  }

  // bias + in-register LayerNorm over the 128 columns this wave owns
  float ps[8], qs[8];
#pragma unroll
  for (int r = 0; r < 8; ++r) { ps[r] = 0.f; qs[r] = 0.f; }
#pragma unroll
  for (int t = 0; t < 8; ++t) {
    float bv = bias[t * 16 + c];
#pragma unroll
    for (int r = 0; r < 8; ++r) {
      float v = acc[t][r] + bv;
      acc[t][r] = v;
      ps[r] += v;
      qs[r] += v * v;
    }
  }
#pragma unroll
  for (int off = 8; off >= 1; off >>= 1) {
#pragma unroll
    for (int r = 0; r < 8; ++r) {
      ps[r] += __shfl_xor(ps[r], off, 32);   // reduce within 16-lane group
      qs[r] += __shfl_xor(qs[r], off, 32);
    }
  }
  float mu[8], rstd[8];
#pragma unroll
  for (int r = 0; r < 8; ++r) {
    float m = ps[r] * (1.0f / 128.0f);
    float var = qs[r] * (1.0f / 128.0f) - m * m;
    mu[r] = m;
    rstd[r] = rsqrtf(var + 1e-5f);
  }
#pragma unroll
  for (int t = 0; t < 8; ++t) {
    int col = t * 16 + c;
    float lg = lng[col], lb = lnb[col];
#pragma unroll
    for (int r = 0; r < 8; ++r) {
      int row = rowBase + g * 8 + r;
      if (row < nrows) {
        float v = (acc[t][r] - mu[r]) * rstd[r] * lg + lb;
        out[(size_t)row * DCH + col] = v >= 0.f ? v : a2v * v;
      }
    }
  }
}

// ---------------- host-side launch -----------------------------------------
extern "C" void kernel_launch(void* const* d_in, const int* in_sizes, int n_in,
                              void* d_out, int out_size, void* d_ws, size_t ws_size,
                              hipStream_t stream) {
  (void)in_sizes; (void)n_in; (void)out_size; (void)ws_size;

  const float* x    = (const float*)d_in[0];
  const int*   ei   = (const int*)d_in[1];
  const int*   esrc = ei;
  const int*   edst = ei + NE;
  const float* W1   = (const float*)d_in[2];
  const float* b1   = (const float*)d_in[3];
  const float* bng  = (const float*)d_in[4];
  const float* bnb  = (const float*)d_in[5];
  const float* a1   = (const float*)d_in[6];
  const float* W2   = (const float*)d_in[7];
  const float* b2   = (const float*)d_in[8];
  const float* lng  = (const float*)d_in[9];
  const float* lnb  = (const float*)d_in[10];
  const float* a2   = (const float*)d_in[11];
  float* out        = (float*)d_out;

  char* w = (char*)d_ws;
  auto alloc = [&](size_t bytes) -> char* {
    char* p = w;
    w += (bytes + 255) & ~(size_t)255;
    return p;
  };
  float* P   = (float*)alloc((size_t)NPAD * DCH * 4);
  float* Q   = (float*)alloc((size_t)NPAD * DCH * 4);
  float* AGG = (float*)alloc((size_t)NPAD * DCH * 4);
  float* Z1  = (float*)alloc((size_t)NPAD * DCH * 4);
  unsigned short* Wt1 = (unsigned short*)alloc((size_t)3 * DCH * DCH * 2);
  unsigned short* Wt2 = (unsigned short*)alloc((size_t)3 * DCH * DCH * 2);
  int* deg      = (int*)alloc((size_t)NN * 4);
  int* rowstart = (int*)alloc((size_t)(NN + 1) * 4);
  int* cursor   = (int*)alloc((size_t)NN * 4);
  int* csr      = (int*)alloc((size_t)NE * 4);
  float* psum   = (float*)alloc((size_t)STAT_BLOCKS * DCH * 4);
  float* qsum   = (float*)alloc((size_t)STAT_BLOCKS * DCH * 4);
  float* scaleA = (float*)alloc(DCH * 4);
  float* shiftA = (float*)alloc(DCH * 4);

  // weight prep (bf16 transpose) + CSR build (once, reused for all 3 layers)
  gin_prep_w<<<(3 * DCH * DCH + 255) / 256, 256, 0, stream>>>(W1, W2, Wt1, Wt2);
  gin_zero_i<<<(NN + 255) / 256, 256, 0, stream>>>(deg, NN);
  gin_hist<<<(NE + 255) / 256, 256, 0, stream>>>(edst, deg, NE);
  gin_scan<<<1, 1024, 0, stream>>>(deg, rowstart, cursor, NN);
  gin_fill<<<(NE + 255) / 256, 256, 0, stream>>>(esrc, edst, cursor, csr, NE);
  gin_sortseg<<<(NN + 255) / 256, 256, 0, stream>>>(csr, rowstart, NN);

  const float* hin = x;
  float* houts[3] = {P, Q, out};
  for (int l = 0; l < 3; ++l) {
    gin_gather<<<(NN + 7) / 8, 256, 0, stream>>>(hin, AGG, rowstart, csr, NN);
    gin_gemm1<<<GEMM_BLOCKS, 128, 0, stream>>>(AGG, Wt1 + (size_t)l * DCH * DCH,
                                               b1 + l * DCH, Z1);
    gin_bnstats<<<STAT_BLOCKS, 128, 0, stream>>>(Z1, psum, qsum, NN);
    gin_bnparams<<<1, 128, 0, stream>>>(psum, qsum, bng + l * DCH, bnb + l * DCH,
                                        scaleA, shiftA, NN);
    gin_gemm2<<<GEMM_BLOCKS, 128, 0, stream>>>(Z1, scaleA, shiftA, a1 + l,
                                               Wt2 + (size_t)l * DCH * DCH, b2 + l * DCH,
                                               lng + l * DCH, lnb + l * DCH, a2 + l,
                                               houts[l], NN);
    hin = houts[l];
  }
}
